// LandmarkDetection_541165879417
// MI455X (gfx1250) — compile-verified, matrix operand order checked by reading, not run on hardware
//
#include <hip/hip_runtime.h>
#include <hip/hip_bf16.h>
#include <math.h>

// ---------------------------------------------------------------------------
// CDNA5 (gfx1250) implementation.
// Core: implicit-GEMM convolution on V_WMMA_F32_16X16X32_F16 (wave32),
// f16 activations/weights, f32 accumulation, async global->LDS weight staging.
// ---------------------------------------------------------------------------

typedef __attribute__((ext_vector_type(16))) _Float16 v16h;
typedef __attribute__((ext_vector_type(8)))  float    v8f;

#define TM  128  // block tile: output channels (8 waves x 16 rows)
#define TNB 64   // block tile: spatial positions (4 WMMA n-tiles per wave)
#define TK  32   // K step (matches 16x16x32 f16 WMMA)

// --- CDNA5 async global->LDS copy (16B per lane) + completion wait ---------
__device__ __forceinline__ void async_copy_b128(const void* gptr, void* lptr)
{
    // VGLOBAL async load: VDST = LDS byte address, VADDR = 64-bit global addr.
    // Flat addresses of __shared__ truncate to the LDS offset (ISA 10.2).
    unsigned           lds = (unsigned)(unsigned long long)lptr;
    unsigned long long g   = (unsigned long long)gptr;
    asm volatile("global_load_async_to_lds_b128 %0, %1, off"
                 :: "v"(lds), "v"(g) : "memory");
}
__device__ __forceinline__ void wait_async0()
{
    asm volatile("s_wait_asynccnt 0" ::: "memory");
}

// Conv as GEMM: M = Cout, N = Hout*Wout (per batch), K = Cin*KH*KW.
// Fused: y = conv(x,W)*scale + bias  (+resid) (ReLU) (+=Y for ASPP accumulate)
__global__ __launch_bounds__(256)
void conv_wmma_kernel(const _Float16* __restrict__ X, const _Float16* __restrict__ Wt,
                      const float* __restrict__ scale, const float* __restrict__ bias,
                      const _Float16* __restrict__ resid, _Float16* __restrict__ Y,
                      int Cin, int Hin, int Win, int Cout, int Hout, int Wout,
                      int KH, int KW, int stride, int pad, int dil,
                      int relu, int accum)
{
    __shared__ _Float16 As[2][TM  * TK] __attribute__((aligned(128)));
    __shared__ _Float16 Bs[2][TNB * TK] __attribute__((aligned(128)));

    const int tid  = threadIdx.x;
    const int wave = tid >> 5;      // 0..7
    const int lane = tid & 31;
    const int HW   = Hout * Wout;
    const int K    = Cin * KH * KW;
    const int KHW  = KH * KW;

    const int bm = blockIdx.y * TM;    // out-channel base
    const int bn = blockIdx.x * TNB;   // spatial base
    const int z  = blockIdx.z;         // batch

    const _Float16* Xb = X + (size_t)z * Cin * Hin * Win;
    _Float16*       Yb = Y + (size_t)z * Cout * HW;
    const _Float16* Rb = resid ? (resid + (size_t)z * Cout * HW) : (const _Float16*)0;

    v8f acc[4];
    #pragma unroll
    for (int j = 0; j < 4; ++j) acc[j] = (v8f){0.f,0.f,0.f,0.f,0.f,0.f,0.f,0.f};

    // ---- cooperative staging of one K-slab into LDS ----
    auto stage = [&](int kk, int buf) {
        // A tile: weights TM x TK, K contiguous per row (OIHW) -> async b128.
        // 128*32 f16 = 8KB = 512 x 16B chunks, 2 per thread.
        #pragma unroll 2
        for (int c = tid; c < (TM * TK) / 8; c += 256) {
            int m  = c >> 2;            // 4 chunks per 32-wide row
            int k8 = (c & 3) * 8;
            int co = bm + m, kg = kk + k8;
            if (co < Cout && kg + 8 <= K) {
                async_copy_b128(Wt + (size_t)co * K + kg, &As[buf][m * TK + k8]);
            } else {
                #pragma unroll
                for (int t = 0; t < 8; ++t) {
                    int kgt = kg + t;
                    _Float16 v = (_Float16)0.f;
                    if (co < Cout && kgt < K) v = Wt[(size_t)co * K + kgt];
                    As[buf][m * TK + k8 + t] = v;
                }
            }
        }
        // B tile: implicit im2col gather TNB x TK (row = spatial position).
        // 64*32 = 2048 elements, 8 per thread.
        #pragma unroll 4
        for (int i = tid; i < TNB * TK; i += 256) {
            int n = i >> 5, k = i & (TK - 1);
            int p = bn + n, kg = kk + k;
            _Float16 v = (_Float16)0.f;
            if (p < HW && kg < K) {
                int ci = kg / KHW, r = kg - ci * KHW;
                int kh = r / KW,  kw = r - kh * KW;
                int oy = p / Wout, ox = p - oy * Wout;
                int iy = oy * stride - pad + kh * dil;
                int ix = ox * stride - pad + kw * dil;
                if ((unsigned)iy < (unsigned)Hin && (unsigned)ix < (unsigned)Win)
                    v = Xb[((size_t)ci * Hin + iy) * Win + ix];
            }
            Bs[buf][n * TK + k] = v;
        }
    };

    stage(0, 0);
    wait_async0();
    __syncthreads();

    // Fragment addressing (contiguous k order, used for both A and B):
    // lanes 0-15 hold K=0..15 contiguous, lanes 16-31 hold K=16..31.
    const int half = lane >> 4;
    const int row  = lane & 15;

    for (int kk = 0; kk < K; kk += TK) {
        const int buf = (kk / TK) & 1;
        if (kk + TK < K) {
            if (lane == 0)   // speculative prefetch of a future weight line
                __builtin_prefetch(Wt + (size_t)(bm + wave * 16) * K + kk + 2 * TK, 0, 0);
            stage(kk + TK, buf ^ 1);
        }
        v16h a = *(const v16h*)&As[buf][(wave * 16 + row) * TK + half * 16];
        #pragma unroll
        for (int j = 0; j < 4; ++j) {
            v16h b = *(const v16h*)&Bs[buf][(j * 16 + row) * TK + half * 16];
            // D = A x B + C  (16x16x32 f16, f32 accumulate); A reused 4x.
            acc[j] = __builtin_amdgcn_wmma_f32_16x16x32_f16(
                false, a, false, b, (short)0, acc[j], false, false);
        }
        wait_async0();
        __syncthreads();
    }

    // ---- epilogue. C/D layout: VGPR r -> M = r (lanes 0-15) / r+8 (16-31),
    // N = lane & 15.
    const int mbase = bm + wave * 16 + ((lane >> 4) ? 8 : 0);
    #pragma unroll
    for (int j = 0; j < 4; ++j) {
        int n = bn + j * 16 + (lane & 15);
        if (n < HW) {
            #pragma unroll
            for (int r = 0; r < 8; ++r) {
                int co = mbase + r;
                if (co < Cout) {
                    float v = acc[j][r];
                    float s  = scale ? scale[co] : 1.f;
                    float bb = bias  ? bias[co]  : 0.f;
                    v = v * s + bb;
                    size_t off = (size_t)co * HW + n;
                    if (Rb)    v += (float)Rb[off];
                    if (relu)  v = fmaxf(v, 0.f);
                    if (accum) v += (float)Yb[off];
                    Yb[off] = (_Float16)v;
                }
            }
        }
    }
}

// ---------------------------------------------------------------------------
// Elementwise / pooling / resampling helpers (f16 activations)
// ---------------------------------------------------------------------------

__global__ void f32_to_f16_kernel(const float* __restrict__ in, _Float16* __restrict__ out,
                                  size_t n)
{
    for (size_t i = (size_t)blockIdx.x * blockDim.x + threadIdx.x; i < n;
         i += (size_t)gridDim.x * blockDim.x)
        out[i] = (_Float16)in[i];
}

__global__ void repeat3_kernel(const float* __restrict__ in, _Float16* __restrict__ out,
                               size_t HW)
{
    for (size_t i = (size_t)blockIdx.x * blockDim.x + threadIdx.x; i < HW;
         i += (size_t)gridDim.x * blockDim.x) {
        _Float16 v = (_Float16)in[i];
        out[i] = v; out[HW + i] = v; out[2 * HW + i] = v;
    }
}

__global__ void maxpool3x3s2_kernel(const _Float16* __restrict__ in, _Float16* __restrict__ out,
                                    int C, int H, int W, int Ho, int Wo)
{
    size_t total = (size_t)C * Ho * Wo;
    for (size_t i = (size_t)blockIdx.x * blockDim.x + threadIdx.x; i < total;
         i += (size_t)gridDim.x * blockDim.x) {
        int wo = (int)(i % Wo); size_t t = i / Wo;
        int ho = (int)(t % Ho); int c = (int)(t / Ho);
        const _Float16* p = in + (size_t)c * H * W;
        float m = -3.4e38f;
        #pragma unroll
        for (int ky = 0; ky < 3; ++ky) {
            int y = ho * 2 - 1 + ky;
            if ((unsigned)y >= (unsigned)H) continue;
            #pragma unroll
            for (int kx = 0; kx < 3; ++kx) {
                int x = wo * 2 - 1 + kx;
                if ((unsigned)x >= (unsigned)W) continue;
                m = fmaxf(m, (float)p[(size_t)y * W + x]);
            }
        }
        out[i] = (_Float16)m;
    }
}

// align-corners bilinear upsample by factor s, optional fused skip add
template <typename TO>
__global__ void upsample_ac_kernel(const _Float16* __restrict__ in,
                                   const _Float16* __restrict__ skip,
                                   TO* __restrict__ out,
                                   int C, int H, int W, int s)
{
    int Ho = H * s, Wo = W * s;
    size_t total = (size_t)C * Ho * Wo;
    for (size_t i = (size_t)blockIdx.x * blockDim.x + threadIdx.x; i < total;
         i += (size_t)gridDim.x * blockDim.x) {
        int wo = (int)(i % Wo); size_t t = i / Wo;
        int ho = (int)(t % Ho); int c = (int)(t / Ho);
        float py = (H > 1) ? ho * (float)(H - 1) / (float)(Ho - 1) : 0.f;
        float px = (W > 1) ? wo * (float)(W - 1) / (float)(Wo - 1) : 0.f;
        int y0 = (int)py; int y1 = y0 + 1; if (y1 > H - 1) y1 = H - 1;
        int x0 = (int)px; int x1 = x0 + 1; if (x1 > W - 1) x1 = W - 1;
        float wy = py - (float)y0, wx = px - (float)x0;
        const _Float16* p = in + (size_t)c * H * W;
        float v = ((float)p[(size_t)y0 * W + x0] * (1.f - wy) + (float)p[(size_t)y1 * W + x0] * wy) * (1.f - wx)
                + ((float)p[(size_t)y0 * W + x1] * (1.f - wy) + (float)p[(size_t)y1 * W + x1] * wy) * wx;
        if (skip) v += (float)skip[i];
        out[i] = (TO)v;
    }
}

// single-block iterative top-44 (mutates its input copy)
__global__ void topk44_kernel(_Float16* __restrict__ heat, int HW, int Wd,
                              int* __restrict__ coords)
{
    __shared__ float sv[256];
    __shared__ int   si[256];
    const int tid = threadIdx.x;
    for (int t = 0; t < 44; ++t) {
        float best = -3.4e38f; int bi = 0;
        for (int i = tid; i < HW; i += 256) {
            float v = (float)heat[i];
            if (v > best) { best = v; bi = i; }
        }
        sv[tid] = best; si[tid] = bi;
        __syncthreads();
        for (int st = 128; st > 0; st >>= 1) {
            if (tid < st && sv[tid + st] > sv[tid]) { sv[tid] = sv[tid + st]; si[tid] = si[tid + st]; }
            __syncthreads();
        }
        if (tid == 0) {
            int idx = si[0];
            heat[idx] = (_Float16)(-60000.f);
            coords[t * 2 + 0] = idx % Wd;   // xc
            coords[t * 2 + 1] = idx / Wd;   // yc
        }
        __syncthreads();
    }
}

// extract 44 zero-padded 96x96 patches, replicated to 3 channels
__global__ void extract_patches_kernel(const float* __restrict__ img, int H, int W,
                                       const int* __restrict__ coords,
                                       _Float16* __restrict__ out)
{
    const int pid = blockIdx.x;                 // 0..43
    const int xc = coords[pid * 2 + 0];
    const int yc = coords[pid * 2 + 1];
    for (int i = threadIdx.x; i < 96 * 96; i += blockDim.x) {
        int py = i / 96, px = i % 96;
        int iy = xc + py - 48;                  // reference slices padded[c0+i][c1+j]
        int ix = yc + px - 48;
        float v = ((unsigned)iy < (unsigned)H && (unsigned)ix < (unsigned)W)
                      ? img[(size_t)iy * W + ix] : 0.f;
        #pragma unroll
        for (int c = 0; c < 3; ++c)
            out[(((size_t)pid * 3 + c) * 96 + py) * 96 + px] = (_Float16)v;
    }
}

// ---------------------------------------------------------------------------
// Host-side orchestration
// ---------------------------------------------------------------------------

namespace {

struct BNp  { const float* s; const float* b; };
struct BlkH { const _Float16* w1; BNp bn1; const _Float16* w2; BNp bn2;
              const _Float16* dw; BNp dbn; };
struct ResH { const _Float16* c1; BNp bn1; BlkH blk[4][2]; };
struct WBh  { const _Float16* w; const float* b; };

inline unsigned ew_blocks(size_t total) {
    size_t b = (total + 255) / 256;
    if (b > (1u << 18)) b = (1u << 18);
    return (unsigned)b;
}

template <typename F>
ResH read_res_h(void* const* d_in, int& i, F&& toh) {
    auto nxf = [&]() -> const float* { return (const float*)d_in[i++]; };
    ResH r;
    r.c1 = toh(nxf(), (size_t)64 * 3 * 7 * 7);
    r.bn1 = { nxf(), nxf() };
    int cin = 64;
    const int couts[4] = { 64, 128, 256, 512 };
    for (int l = 0; l < 4; ++l) {
        const int cout = couts[l];
        for (int b = 0; b < 2; ++b) {
            const int cinb = (b == 0) ? cin : cout;
            BlkH k;
            k.w1 = toh(nxf(), (size_t)cout * cinb * 9); k.bn1 = { nxf(), nxf() };
            k.w2 = toh(nxf(), (size_t)cout * cout * 9); k.bn2 = { nxf(), nxf() };
            if (l > 0 && b == 0) { k.dw = toh(nxf(), (size_t)cout * cinb); k.dbn = { nxf(), nxf() }; }
            else                 { k.dw = nullptr; k.dbn = { nullptr, nullptr }; }
            r.blk[l][b] = k;
        }
        cin = cout;
    }
    return r;
}

void conv(hipStream_t st, const _Float16* X, const _Float16* Wt,
          const float* sc, const float* bi, const _Float16* resid, _Float16* Y,
          int N, int Cin, int H, int W, int Cout, int KH, int KW,
          int s, int pad, int dil, int relu, int accum, int& Ho, int& Wo)
{
    Ho = (H + 2 * pad - dil * (KH - 1) - 1) / s + 1;
    Wo = (W + 2 * pad - dil * (KW - 1) - 1) / s + 1;
    dim3 grid((unsigned)((Ho * Wo + TNB - 1) / TNB), (unsigned)((Cout + TM - 1) / TM), (unsigned)N);
    conv_wmma_kernel<<<grid, 256, 0, st>>>(X, Wt, sc, bi, resid, Y,
                                           Cin, H, W, Cout, Ho, Wo,
                                           KH, KW, s, pad, dil, relu, accum);
}

// full ResNet-18 feature stack (batch N in blockIdx.z / channel folding)
void resnet_fwd(hipStream_t st, const _Float16* x3, int N, int H, int W, const ResH& r,
                _Float16* c1buf, _Float16* poolbuf, _Float16* t1, _Float16* tsc,
                _Float16* acttmp, _Float16* louts[4], int Hs[4], int Ws[4])
{
    int Ho, Wo;
    conv(st, x3, r.c1, r.bn1.s, r.bn1.b, nullptr, c1buf,
         N, 3, H, W, 64, 7, 7, 2, 3, 1, /*relu*/1, 0, Ho, Wo);
    int Hp = (Ho + 2 - 3) / 2 + 1, Wp = (Wo + 2 - 3) / 2 + 1;
    size_t pt = (size_t)N * 64 * Hp * Wp;
    maxpool3x3s2_kernel<<<ew_blocks(pt), 256, 0, st>>>(c1buf, poolbuf, N * 64, Ho, Wo, Hp, Wp);

    const _Float16* cur = poolbuf;
    int Hc = Hp, Wc = Wp, Cin = 64;
    const int Couts[4]   = { 64, 128, 256, 512 };
    const int strides[4] = { 1, 2, 2, 2 };
    for (int l = 0; l < 4; ++l) {
        const int Cout = Couts[l];
        for (int b = 0; b < 2; ++b) {
            const BlkH& k = r.blk[l][b];
            int s = (b == 0) ? strides[l] : 1;
            _Float16* out = (b == 1) ? louts[l] : acttmp;
            int H1, W1, H2, W2, hh, ww;
            conv(st, cur, k.w1, k.bn1.s, k.bn1.b, nullptr, t1,
                 N, Cin, Hc, Wc, Cout, 3, 3, s, 1, 1, /*relu*/1, 0, H1, W1);
            const _Float16* sc = cur;
            if (k.dw) {
                conv(st, cur, k.dw, k.dbn.s, k.dbn.b, nullptr, tsc,
                     N, Cin, Hc, Wc, Cout, 1, 1, s, 0, 1, /*relu*/0, 0, hh, ww);
                sc = tsc;
            }
            conv(st, t1, k.w2, k.bn2.s, k.bn2.b, sc, out,
                 N, Cout, H1, W1, Cout, 3, 3, 1, 1, 1, /*relu*/1, 0, H2, W2);
            cur = out; Hc = H2; Wc = W2; Cin = Cout;
        }
        Hs[l] = Hc; Ws[l] = Wc;
    }
}

} // namespace

extern "C" void kernel_launch(void* const* d_in, const int* in_sizes, int n_in,
                              void* d_out, int out_size, void* d_ws, size_t ws_size,
                              hipStream_t stream)
{
    (void)in_sizes; (void)n_in; (void)out_size; (void)ws_size;
    static const int DIL[4] = { 1, 6, 12, 18 };

    // ---- workspace bump allocator ----
    char* wsb = (char*)d_ws;
    size_t off = 0;
    auto allocb = [&](size_t bytes) -> void* {
        void* p = (void*)(wsb + off);
        off += (bytes + 255) & ~(size_t)255;
        return p;
    };
    auto fa = [&](size_t elems) -> _Float16* { return (_Float16*)allocb(elems * 2); };

    // weight f32 -> f16 conversion (each launch; deterministic)
    auto toh = [&](const float* src, size_t n) -> const _Float16* {
        _Float16* p = fa(n);
        f32_to_f16_kernel<<<ew_blocks(n), 256, 0, stream>>>(src, p, n);
        return p;
    };

    // ---- walk flattened parameter list (setup_inputs dict order) ----
    int ci = 0;
    const float* x = (const float*)d_in[ci++];           // (1,1,512,512) f32
    ResH g_res = read_res_h(d_in, ci, toh);
    WBh g_dec[3];
    const size_t dec_n[3] = { (size_t)256 * 512, (size_t)128 * 256, (size_t)64 * 128 };
    for (int i = 0; i < 3; ++i) {
        g_dec[i].w = toh((const float*)d_in[ci], dec_n[i]); ci++;
        g_dec[i].b = (const float*)d_in[ci++];
    }
    WBh g_aspp[4];
    for (int i = 0; i < 4; ++i) {
        g_aspp[i].w = toh((const float*)d_in[ci], (size_t)256 * 64 * 9); ci++;
        g_aspp[i].b = (const float*)d_in[ci++];
    }
    WBh g_head = { toh((const float*)d_in[ci], 256), (const float*)d_in[ci + 1] }; ci += 2;
    ResH l_res = read_res_h(d_in, ci, toh);
    WBh l_aspp[4];
    for (int i = 0; i < 4; ++i) {
        l_aspp[i].w = toh((const float*)d_in[ci], (size_t)256 * 512 * 9); ci++;
        l_aspp[i].b = (const float*)d_in[ci++];
    }

    // ---- activation buffers (all f16) ----
    _Float16* inbuf   = fa(1216512);   // max(3*512*512, 44*3*96*96)
    _Float16* c1buf   = fa(6488064);   // max(64*256*256, 44*64*48*48)
    _Float16* poolbuf = fa(1622016);
    _Float16* t1      = fa(1622016);
    _Float16* tsc     = fa(1622016);
    _Float16* acttmp  = fa(1622016);
    _Float16* l1      = fa(1622016);
    _Float16* l2      = fa(811008);
    _Float16* l3      = fa(405504);
    _Float16* l4      = fa(202752);
    _Float16* dectmp  = fa(262144);    // max decoder conv output (64*64*64)
    _Float16* u1      = fa(262144);    // 256*32*32
    _Float16* u2      = fa(524288);    // 128*64*64
    _Float16* u3      = fa(1048576);   // 64*128*128
    _Float16* asppA   = fa(4194304);   // 256*128*128
    _Float16* headsm  = fa(16384);     // 1*128*128
    _Float16* heat    = fa(262144);    // 512*512
    _Float16* heatcp  = fa(262144);
    int*      coords  = (int*)allocb(44 * 2 * sizeof(int));
    _Float16* lasppo  = fa(101376);    // 44*256*3*3

    _Float16* louts[4] = { l1, l2, l3, l4 };
    int Hs[4], Ws[4], Ho, Wo;

    // ================= global branch =================
    repeat3_kernel<<<ew_blocks((size_t)512 * 512), 256, 0, stream>>>(x, inbuf, (size_t)512 * 512);
    resnet_fwd(stream, inbuf, 1, 512, 512, g_res, c1buf, poolbuf, t1, tsc, acttmp, louts, Hs, Ws);
    // dims: l1 64@128^2, l2 128@64^2, l3 256@32^2, l4 512@16^2

    // decoder: u1 = up2(1x1(l4)) + l3 ; u2 = up2(1x1(u1)) + l2 ; u3 = up2(1x1(u2)) + l1
    conv(stream, l4, g_dec[0].w, nullptr, g_dec[0].b, nullptr, dectmp,
         1, 512, Hs[3], Ws[3], 256, 1, 1, 1, 0, 1, 0, 0, Ho, Wo);
    upsample_ac_kernel<_Float16><<<ew_blocks((size_t)256 * Hs[2] * Ws[2]), 256, 0, stream>>>(
        dectmp, l3, u1, 256, Ho, Wo, 2);
    conv(stream, u1, g_dec[1].w, nullptr, g_dec[1].b, nullptr, dectmp,
         1, 256, Hs[2], Ws[2], 128, 1, 1, 1, 0, 1, 0, 0, Ho, Wo);
    upsample_ac_kernel<_Float16><<<ew_blocks((size_t)128 * Hs[1] * Ws[1]), 256, 0, stream>>>(
        dectmp, l2, u2, 128, Ho, Wo, 2);
    conv(stream, u2, g_dec[2].w, nullptr, g_dec[2].b, nullptr, dectmp,
         1, 128, Hs[1], Ws[1], 64, 1, 1, 1, 0, 1, 0, 0, Ho, Wo);
    upsample_ac_kernel<_Float16><<<ew_blocks((size_t)64 * Hs[0] * Ws[0]), 256, 0, stream>>>(
        dectmp, l1, u3, 64, Ho, Wo, 2);

    // global ASPP: sum of 4 dilated 3x3 convs (64 -> 256) at 128x128
    for (int i = 0; i < 4; ++i)
        conv(stream, u3, g_aspp[i].w, nullptr, g_aspp[i].b, nullptr, asppA,
             1, 64, Hs[0], Ws[0], 256, 3, 3, 1, DIL[i], DIL[i], 0, /*accum*/ i > 0, Ho, Wo);

    // head: 1x1 (256 -> 1), up4 -> 512x512 heatmap, then top-44
    conv(stream, asppA, g_head.w, nullptr, g_head.b, nullptr, headsm,
         1, 256, 128, 128, 1, 1, 1, 1, 0, 1, 0, 0, Ho, Wo);
    upsample_ac_kernel<_Float16><<<ew_blocks((size_t)512 * 512), 256, 0, stream>>>(
        headsm, nullptr, heat, 1, 128, 128, 4);
    hipMemcpyAsync(heatcp, heat, (size_t)512 * 512 * sizeof(_Float16),
                   hipMemcpyDeviceToDevice, stream);
    topk44_kernel<<<1, 256, 0, stream>>>(heatcp, 512 * 512, 512, coords);

    // ================= local branch =================
    extract_patches_kernel<<<44, 256, 0, stream>>>(x, 512, 512, coords, inbuf);
    resnet_fwd(stream, inbuf, 44, 96, 96, l_res, c1buf, poolbuf, t1, tsc, acttmp, louts, Hs, Ws);
    // l4 local: (44, 512, 3, 3)

    // local ASPP: sum of 4 dilated 3x3 convs (512 -> 256) at 3x3, batch 44
    for (int i = 0; i < 4; ++i)
        conv(stream, l4, l_aspp[i].w, nullptr, l_aspp[i].b, nullptr, lasppo,
             44, 512, Hs[3], Ws[3], 256, 3, 3, 1, DIL[i], DIL[i], 0, /*accum*/ i > 0, Ho, Wo);

    // final 4x align-corners upsample (3x3 -> 12x12) straight into d_out (f32)
    upsample_ac_kernel<float><<<ew_blocks((size_t)44 * 256 * 12 * 12), 256, 0, stream>>>(
        lasppo, nullptr, (float*)d_out, 44 * 256, 3, 3, 4);
}
